// AdaptiveRouter_25898652795233
// MI455X (gfx1250) — compile-verified, matrix operand order checked by reading, not run on hardware
//
#include <hip/hip_runtime.h>

typedef __attribute__((ext_vector_type(16))) __bf16 v16bf;
typedef __attribute__((ext_vector_type(8)))  float  v8f;

#define T_TOK   16384
#define DDIM    2048
#define NEXP    64
#define BLOCK_M 64          // tokens per workgroup (4 waves x 16)
#define KT      32          // bf16 WMMA K per step
#define NCHUNK  (DDIM / KT) // 64 K-chunks
#define XSTR    36          // padded f32 row stride: 144B = 16B-aligned, bank-conflict-free
#define BSTR    40          // fallback bf16 row stride

// B-fragment store: [chunk][tile][quad][lane] * 8 bf16 (16B) each
#define BFRAG_BYTES ((size_t)NCHUNK * 4 * 2 * 32 * 16)   // 256 KB
#define WS_ACC_BYTES 256

// Pack two f32 into packed bf16 with ONE v_perm_b32 (take high 16 bits of each).
// RTZ instead of RNE: sub-LSB vs bf16 quantization noise, fine for routing.
static __device__ __forceinline__ unsigned pk2(float a, float b) {
  unsigned au = __builtin_bit_cast(unsigned, a);
  unsigned bu = __builtin_bit_cast(unsigned, b);
  // S0 = bu (bytes 4..7), S1 = au (bytes 0..3); D = {b.hi16, a.hi16}
  return __builtin_amdgcn_perm(bu, au, 0x07060302u);
}
static __device__ __forceinline__ unsigned short f2bf(float f) {
  return (unsigned short)(__builtin_bit_cast(unsigned, f) >> 16);
}

// ---- prep: gate_w f32 [D][E] -> bf16 B-fragments in WMMA lane order ----
__global__ __launch_bounds__(256)
void prep_bfrag(const float* __restrict__ gw, unsigned short* __restrict__ bp) {
  int id   = blockIdx.x * 256 + threadIdx.x;  // 0 .. 16383 segments of 8 bf16
  int lane = id & 31;
  int q    = (id >> 5) & 1;
  int t    = (id >> 6) & 3;
  int c    = id >> 8;
  int n    = t * 16 + (lane & 15);
  int g    = lane >> 4;
  int kb   = c * KT + g * 8 + (q ? 16 : 0);   // ISA 7.12.2 16-bit K halves
  unsigned u[4];
#pragma unroll
  for (int j = 0; j < 4; ++j)
    u[j] = pk2(gw[(size_t)(kb + 2 * j) * NEXP + n],
               gw[(size_t)(kb + 2 * j + 1) * NEXP + n]);
  *(uint4*)(bp + (size_t)id * 8) = make_uint4(u[0], u[1], u[2], u[3]);
}

template <bool PREPPED>
__global__ __launch_bounds__(128)
void router_gemm_topk(const float* __restrict__ x,
                      const float* __restrict__ gw,
                      const float* __restrict__ loads,
                      float* __restrict__ out,
                      float* __restrict__ ws_acc,
                      const unsigned short* __restrict__ ws_b) {
  __shared__ __align__(16) float lds_x[2][BLOCK_M * XSTR];       // double-buffered async DMA
  __shared__ __align__(16) unsigned short lds_b[NEXP * BSTR];    // fallback path only
  __shared__ float lds_logits[BLOCK_M][NEXP + 1];
  __shared__ float lds_bias[NEXP];
  __shared__ float lds_loads[NEXP];

  const int tid  = threadIdx.x;
  const int wave = tid >> 5;
  const int lane = tid & 31;
  const int row0 = blockIdx.x * BLOCK_M;
  const int m    = lane & 15;
  const int g    = lane >> 4;

  if (tid < NEXP) {
    lds_bias[tid]  = -(loads[tid] - (1.0f / NEXP)) * 2.0f;
    lds_loads[tid] = 0.0f;
  }

  // issue one x-tile (64x32 f32) as 4x 16B async-to-LDS copies per thread
  auto issue_x = [&](int buf, int k0) {
#pragma unroll
    for (int i = 0; i < 4; ++i) {
      int e = (i * 128 + tid) * 4;
      int r = e >> 5, col = e & 31;
      unsigned lds_off = (unsigned)(uintptr_t)(&lds_x[buf][r * XSTR + col]);
      const float* gp = x + (size_t)(row0 + r) * DDIM + k0 + col;
      asm volatile("global_load_async_to_lds_b128 %0, %1, off"
                   :: "v"(lds_off), "v"(gp) : "memory");
    }
  };

  v8f acc[4] = {};
  issue_x(0, 0);  // prologue

  for (int c = 0; c < NCHUNK; ++c) {
    const int cur = c & 1;
    __syncthreads();  // A: prior reads of buf[cur^1] complete before overwrite
    if (c + 1 < NCHUNK) issue_x(cur ^ 1, (c + 1) * KT);
    if constexpr (!PREPPED) {
      // fallback: stage gate_w chunk, transposed to [expert][k] bf16
#pragma unroll
      for (int i = 0; i < 16; ++i) {
        int linear = i * 128 + tid;
        int n = linear & 63;
        int k = linear >> 6;
        lds_b[n * BSTR + k] = f2bf(gw[(size_t)(c * KT + k) * NEXP + n]);
      }
    }
    if (c + 1 < NCHUNK)
      asm volatile("s_wait_asynccnt 0x4" ::: "memory");  // in-order: first 4 (buf[cur]) done
    else
      asm volatile("s_wait_asynccnt 0x0" ::: "memory");
    __syncthreads();  // B: buf[cur] visible to all waves

    // ---- A fragment: 16 f32 from LDS -> 8x v_perm_b32 into v16bf ----
    const float* xr = &lds_x[cur][(wave * 16 + m) * XSTR];
    float4 q0 = *(const float4*)(xr + g * 8);
    float4 q1 = *(const float4*)(xr + g * 8 + 4);
    float4 q2 = *(const float4*)(xr + 16 + g * 8);
    float4 q3 = *(const float4*)(xr + 16 + g * 8 + 4);
    union { unsigned u[8]; v16bf v; } af;
    af.u[0] = pk2(q0.x, q0.y); af.u[1] = pk2(q0.z, q0.w);
    af.u[2] = pk2(q1.x, q1.y); af.u[3] = pk2(q1.z, q1.w);
    af.u[4] = pk2(q2.x, q2.y); af.u[5] = pk2(q2.z, q2.w);
    af.u[6] = pk2(q3.x, q3.y); af.u[7] = pk2(q3.z, q3.w);

#pragma unroll
    for (int t = 0; t < 4; ++t) {
      union { uint4 q[2]; v16bf v; } bf_;
      if constexpr (PREPPED) {
        const unsigned short* bp =
            ws_b + ((size_t)(((c * 4 + t) * 2) * 32 + lane)) * 8;
        bf_.q[0] = *(const uint4*)(bp);
        bf_.q[1] = *(const uint4*)(bp + 32 * 8);  // q=1 segment
      } else {
        const unsigned short* brow = &lds_b[(t * 16 + m) * BSTR];
        bf_.q[0] = *(const uint4*)(brow + g * 8);
        bf_.q[1] = *(const uint4*)(brow + 16 + g * 8);
      }
      acc[t] = __builtin_amdgcn_wmma_f32_16x16x32_bf16(
          false, af.v, false, bf_.v, (short)0, acc[t], false, false);
    }
  }

  __syncthreads();
  // ---- spill logits per C/D layout: VGPR r -> rows (g*8 + r), N = t*16 + m ----
#pragma unroll
  for (int t = 0; t < 4; ++t) {
    int n = t * 16 + m;
#pragma unroll
    for (int r = 0; r < 8; ++r)
      lds_logits[wave * 16 + (g * 8 + r)][n] = acc[t][r];
  }
  __syncthreads();

  // ---- branchless biased top-2, softmax over unbiased logits ----
  if (tid < BLOCK_M) {
    const int tok = row0 + tid;
    float best = -3.402823466e38f, sec = -3.402823466e38f;
    int bi = 0, si = 0;
#pragma unroll 8
    for (int e = 0; e < NEXP; ++e) {
      float lb = lds_logits[tid][e] + lds_bias[e];
      bool gb = lb > best;
      bool gs = lb > sec;
      float ns  = gb ? best : (gs ? lb : sec);
      int   nsi = gb ? bi   : (gs ? e  : si);
      best = gb ? lb : best;
      bi   = gb ? e  : bi;
      sec = ns; si = nsi;
    }
    float l0 = lds_logits[tid][bi];
    float l1 = lds_logits[tid][si];
    float mx = fmaxf(l0, l1);
    float p0 = __expf(l0 - mx);
    float p1 = __expf(l1 - mx);
    float inv = 1.0f / (p0 + p1);
    float w0 = p0 * inv, w1 = p1 * inv;
    out[(size_t)tok * 2 + 0] = w0;
    out[(size_t)tok * 2 + 1] = w1;
    out[(size_t)2 * T_TOK + (size_t)tok * 2 + 0] = (float)bi;
    out[(size_t)2 * T_TOK + (size_t)tok * 2 + 1] = (float)si;
    atomicAdd(&lds_loads[bi], w0);
    atomicAdd(&lds_loads[si], w1);
  }
  __syncthreads();
  if (tid < NEXP) atomicAdd(&ws_acc[tid], lds_loads[tid]);
}

__global__ void router_finalize(const float* __restrict__ loads,
                                const float* __restrict__ ws,
                                float* __restrict__ out) {
  int e = threadIdx.x;
  if (e < NEXP) {
    float cur = ws[e] / (float)T_TOK;
    out[(size_t)4 * T_TOK + e] = 0.9f * loads[e] + 0.1f * cur;
  }
}

extern "C" void kernel_launch(void* const* d_in, const int* in_sizes, int n_in,
                              void* d_out, int out_size, void* d_ws, size_t ws_size,
                              hipStream_t stream) {
  const float* x     = (const float*)d_in[0];
  const float* gw    = (const float*)d_in[1];
  const float* loads = (const float*)d_in[2];
  float* out = (float*)d_out;
  float* ws_acc = (float*)d_ws;
  unsigned short* ws_b = (unsigned short*)((char*)d_ws + WS_ACC_BYTES);

  hipMemsetAsync(d_ws, 0, WS_ACC_BYTES, stream);
  const bool prepped = ws_size >= WS_ACC_BYTES + BFRAG_BYTES;
  if (prepped) {
    prep_bfrag<<<NCHUNK, 256, 0, stream>>>(gw, ws_b);
    router_gemm_topk<true><<<T_TOK / BLOCK_M, 128, 0, stream>>>(
        x, gw, loads, out, ws_acc, ws_b);
  } else {
    router_gemm_topk<false><<<T_TOK / BLOCK_M, 128, 0, stream>>>(
        x, gw, loads, out, ws_acc, ws_b);
  }
  router_finalize<<<1, 64, 0, stream>>>(loads, ws_acc, out);
}
